// KernelAttention_481036337893
// MI455X (gfx1250) — matrix operand order repaired and checked by reading, not compiled, not run on hardware
//
#include <hip/hip_runtime.h>
#include <hip/hip_bf16.h>
#include <math.h>

typedef __attribute__((ext_vector_type(16))) _Float16 v16h;
typedef __attribute__((ext_vector_type(8)))  float    v8f;

// ---------------- constants ----------------
constexpr int CDIM   = 256;
constexpr int CHEADS = 4;
constexpr int CDH    = 64;
constexpr int CN     = 6;
constexpr int CG     = 8;
constexpr int CQ     = 4096;   // 64*64
constexpr float CSCALE = 0.125f;  // 64^-0.5
constexpr float CEPS = 1e-5f;

static __device__ __forceinline__ float wave_sum32(float v) {
    v += __shfl_xor(v, 16, 32);
    v += __shfl_xor(v,  8, 32);
    v += __shfl_xor(v,  4, 32);
    v += __shfl_xor(v,  2, 32);
    v += __shfl_xor(v,  1, 32);
    return v;
}

// ================= transpose (d-major -> q-major) for q slabs + skip =============
// input slab: [256][4096] f32 ; output: [4096][256] f32
__global__ __launch_bounds__(256) void kT_transpose(const float* __restrict__ q,
                                                    const float* __restrict__ skip,
                                                    float* __restrict__ qx,
                                                    float* __restrict__ skipT) {
    int bid   = blockIdx.x;
    int tileQ = bid & 127;         // 4096/32
    int tileD = (bid >> 7) & 7;    // 256/32
    int s     = bid >> 10;         // 0..6 (6 q-frames + skip)
    const float* src;
    float* dst;
    if (s < CN) { src = q + (size_t)s * CDIM * CQ; dst = qx + (size_t)s * CQ * CDIM; }
    else        { src = skip;                      dst = skipT; }
    __shared__ float tile[32][33];
    int tx = threadIdx.x & 31, ty = threadIdx.x >> 5;   // 32 x 8
    #pragma unroll
    for (int j = 0; j < 32; j += 8)
        tile[ty + j][tx] = src[(size_t)(tileD * 32 + ty + j) * CQ + tileQ * 32 + tx];
    __syncthreads();
    #pragma unroll
    for (int j = 0; j < 32; j += 8)
        dst[(size_t)(tileQ * 32 + ty + j) * CDIM + tileD * 32 + tx] = tile[tx][ty + j];
}

// ================= weight pre-swizzle into WMMA B-fragment order =================
// src: f32 [K][N] row-major.  dst: f16 tiles [N/16][K/32][32 lanes][16 halves]
// lane<16: halves = K kt*32+0..15, N = nt*16+lane ; lane>=16: K kt*32+16..31, N = nt*16+(lane-16)
__global__ __launch_bounds__(256) void kW_swizzle(const float* __restrict__ w,
                                                  _Float16* __restrict__ dst,
                                                  int K, int N) {
    int t = blockIdx.x * 256 + threadIdx.x;
    int KT = K >> 5;
    int total = (N >> 4) * KT * 32;
    if (t >= total) return;
    int lane = t & 31;
    int tile = t >> 5;
    int kt = tile % KT;
    int nt = tile / KT;
    int kb = (lane < 16) ? 0 : 16;
    int nn = nt * 16 + (lane & 15);
    #pragma unroll
    for (int h = 0; h < 16; ++h)
        dst[(size_t)t * 16 + h] = (_Float16)w[(size_t)(kt * 32 + kb + h) * N + nn];
}

// ================= fused (LN) + GEMM + epilogue ==================================
enum { EP_F16 = 0, EP_GELU_F32 = 1, EP_SKIP_LN = 2, EP_RESID_LN_T = 3 };

union SmemU {
    _Float16 aT[2][16][32][16];   // A fragments: [mtile][ktile][lane][16 halves] (32 KB)
    float    outS[32][261];       // output staging, padded (33.4 KB)
};

template <int KDIM, bool LN_IN, int MODE, bool NT_IN>
__global__ __launch_bounds__(256) void kGemm(
        const float* __restrict__ inp,                 // [M][KDIM] f32
        const float* __restrict__ lnG, const float* __restrict__ lnB,
        const _Float16* __restrict__ wB,               // preswizzled [N/16][KDIM/32][32][16]
        const float* __restrict__ bias,                // [Ntot]
        int Ntot,
        _Float16* __restrict__ outH,                   // EP_F16
        const float* __restrict__ addend,              // EP_SKIP_LN: skipT, EP_RESID: z
        const float* __restrict__ lnG2, const float* __restrict__ lnB2,
        float* __restrict__ outF) {                    // EP_GELU_F32 / EP_SKIP_LN / EP_RESID
    constexpr int KT  = KDIM / 32;
    constexpr int PER = KDIM / 32;   // f32 elems per lane per row
    __shared__ __align__(32) SmemU sm;

    const int w    = threadIdx.x >> 5;
    const int lane = threadIdx.x & 31;
    const int rowBase = blockIdx.x * 32;

    // ---- stage A: 32 rows, optional LN, convert f16, write fragment layout ----
    #pragma unroll
    for (int rr = 0; rr < 4; ++rr) {
        int mloc = w * 4 + rr;
        const float* src = inp + (size_t)(rowBase + mloc) * KDIM;
        float x[PER];
        float s = 0.f, s2 = 0.f;
        #pragma unroll
        for (int j = 0; j < PER; ++j) {
            float vv;
            if constexpr (NT_IN) vv = __builtin_nontemporal_load(&src[lane * PER + j]);
            else                 vv = src[lane * PER + j];
            x[j] = vv;
            if constexpr (LN_IN) { s += vv; s2 += vv * vv; }
        }
        float mean = 0.f, rstd = 1.f;
        if constexpr (LN_IN) {
            s  = wave_sum32(s);
            s2 = wave_sum32(s2);
            mean = s * (1.0f / KDIM);
            float var = s2 * (1.0f / KDIM) - mean * mean;
            rstd = rsqrtf(var + CEPS);
        }
        #pragma unroll
        for (int grp = 0; grp < PER / 8; ++grp) {
            int k0 = lane * PER + grp * 8;
            int kt = k0 >> 5;
            int kk = k0 & 31;                        // 0, 8, 16, 24
            int ln = (kk >= 16) ? 16 + (mloc & 15) : (mloc & 15);
            int hb = (kk & 8) ? 8 : 0;
            _Float16* dst = &sm.aT[mloc >> 4][kt][ln][hb];
            #pragma unroll
            for (int j = 0; j < 8; ++j) {
                float vv = x[grp * 8 + j];
                if constexpr (LN_IN) vv = (vv - mean) * rstd * lnG[k0 + j] + lnB[k0 + j];
                dst[j] = (_Float16)vv;
            }
        }
    }
    __syncthreads();

    // ---- WMMA GEMM: wave -> 32 rows x 32 cols (2 mt x 2 jt accumulators) ----
    // Each B fragment feeds 2 WMMAs (mt0, mt1): halves B traffic, doubles ILP.
    const int nb  = w * 32;
    const int nbG = blockIdx.y * 256 + nb;
    v8f acc[2][2];
    #pragma unroll
    for (int jt = 0; jt < 2; ++jt) {
        float bv = bias[nbG + jt * 16 + (lane & 15)];
        #pragma unroll
        for (int i = 0; i < 8; ++i) { acc[0][jt][i] = bv; acc[1][jt][i] = bv; }
    }
    #pragma unroll
    for (int kt = 0; kt < KT; ++kt) {
        v16h a0 = *(const v16h*)(&sm.aT[0][kt][lane][0]);
        v16h a1 = *(const v16h*)(&sm.aT[1][kt][lane][0]);
        #pragma unroll
        for (int jt = 0; jt < 2; ++jt) {
            const _Float16* bp = wB + ((((size_t)(nbG / 16 + jt) * KT + kt) * 32 + lane) * 16);
            v16h b = *(const v16h*)bp;
            acc[0][jt] = __builtin_amdgcn_wmma_f32_16x16x32_f16(
                false, a0, false, b, (short)0, acc[0][jt], false, false);
            acc[1][jt] = __builtin_amdgcn_wmma_f32_16x16x32_f16(
                false, a1, false, b, (short)0, acc[1][jt], false, false);
        }
    }
    __syncthreads();   // done reading aT before reusing union as outS

    // ---- spill accumulators to LDS ----
    {
        int rOff  = (lane < 16) ? 0 : 8;
        int cBase = nb + (lane & 15);
        #pragma unroll
        for (int mt = 0; mt < 2; ++mt)
            #pragma unroll
            for (int jt = 0; jt < 2; ++jt)
                #pragma unroll
                for (int i = 0; i < 8; ++i)
                    sm.outS[mt * 16 + rOff + i][cBase + jt * 16] = acc[mt][jt][i];
    }
    __syncthreads();

    // ---- epilogues ----
    if constexpr (MODE == EP_F16) {
        int row = threadIdx.x >> 3;
        int c0  = (threadIdx.x & 7) * 32;
        _Float16* dst = outH + (size_t)(rowBase + row) * Ntot + blockIdx.y * 256 + c0;
        #pragma unroll
        for (int j = 0; j < 32; ++j) dst[j] = (_Float16)sm.outS[row][c0 + j];
    } else if constexpr (MODE == EP_GELU_F32) {
        int row = threadIdx.x >> 3;
        int c0  = (threadIdx.x & 7) * 32;
        float* dst = outF + (size_t)(rowBase + row) * Ntot + blockIdx.y * 256 + c0;
        #pragma unroll
        for (int j = 0; j < 32; ++j) {
            float vv = sm.outS[row][c0 + j];
            dst[j] = 0.5f * vv * (1.0f + erff(vv * 0.70710678118f));
        }
    } else {   // EP_SKIP_LN or EP_RESID_LN_T : per-row add + LayerNorm (Ntot==256)
        #pragma unroll
        for (int rr = 0; rr < 4; ++rr) {
            int row = w * 4 + rr;
            size_t rowG = rowBase + row;
            float x[8];
            float s = 0.f, s2 = 0.f;
            #pragma unroll
            for (int j = 0; j < 8; ++j) {
                int c = lane * 8 + j;
                float vv = sm.outS[row][c] + addend[rowG * CDIM + c];
                x[j] = vv; s += vv; s2 += vv * vv;
            }
            s  = wave_sum32(s);
            s2 = wave_sum32(s2);
            float mean = s * (1.0f / CDIM);
            float var  = s2 * (1.0f / CDIM) - mean * mean;
            float rstd = rsqrtf(var + CEPS);
            #pragma unroll
            for (int j = 0; j < 8; ++j) {
                int c = lane * 8 + j;
                float nv = (x[j] - mean) * rstd * lnG2[c] + lnB2[c];
                if constexpr (MODE == EP_SKIP_LN) outF[rowG * CDIM + c] = nv;
                else                              sm.outS[row][c] = nv;
            }
        }
        if constexpr (MODE == EP_RESID_LN_T) {
            __syncthreads();
            // transposed store: out[d][qpos]; lanes sweep qpos (coalesced 128B)
            for (int dd = w; dd < CDIM; dd += 8)
                outF[(size_t)dd * CQ + rowBase + lane] = sm.outS[lane][dd];
        }
    }
}

// ================= attention: one wave per (qpos, head) ==========================
__global__ __launch_bounds__(256) void kAttn(const _Float16* __restrict__ qp,
                                             const _Float16* __restrict__ kp,
                                             const _Float16* __restrict__ vp,
                                             const unsigned char* __restrict__ mask,
                                             float* __restrict__ aOut) {
    int lane = threadIdx.x & 31;
    int task = blockIdx.x * 8 + (threadIdx.x >> 5);   // 0..16383
    int m    = task & 3;
    int qpos = task >> 2;
    int d0   = lane * 2;

    float sc[CN * CG];
    #pragma unroll
    for (int n = 0; n < CN; ++n) {
        const _Float16* qv = qp + ((size_t)(n * CQ + qpos)) * CDIM + m * CDH + d0;
        float q0 = (float)qv[0], q1 = (float)qv[1];
        bool ok = mask[n * CQ + qpos] != 0;
        #pragma unroll
        for (int g = 0; g < CG; ++g) {
            const _Float16* kv = kp + ((size_t)((n * CQ + qpos) * CG + g)) * CDIM + m * CDH + d0;
            float p = q0 * (float)kv[0] + q1 * (float)kv[1];
            p = wave_sum32(p);
            sc[n * CG + g] = ok ? p * CSCALE : -1e9f;
        }
    }
    float mx = sc[0];
    #pragma unroll
    for (int i = 1; i < CN * CG; ++i) mx = fmaxf(mx, sc[i]);
    float ssum = 0.f;
    #pragma unroll
    for (int i = 0; i < CN * CG; ++i) { sc[i] = __expf(sc[i] - mx); ssum += sc[i]; }
    float inv = 1.0f / ssum;
    float a0 = 0.f, a1 = 0.f;
    #pragma unroll
    for (int n = 0; n < CN; ++n)
        #pragma unroll
        for (int g = 0; g < CG; ++g) {
            const _Float16* vv = vp + ((size_t)((n * CQ + qpos) * CG + g)) * CDIM + m * CDH + d0;
            float wgt = sc[n * CG + g] * inv;
            a0 += wgt * (float)vv[0];
            a1 += wgt * (float)vv[1];
        }
    float* dst = aOut + (size_t)qpos * CDIM + m * CDH + d0;
    dst[0] = a0; dst[1] = a1;
}

// ================= host launcher =================================================
extern "C" void kernel_launch(void* const* d_in, const int* in_sizes, int n_in,
                              void* d_out, int out_size, void* d_ws, size_t ws_size,
                              hipStream_t stream) {
    const float* q      = (const float*)d_in[0];
    const float* k      = (const float*)d_in[1];
    const float* v      = (const float*)d_in[2];
    const float* skip   = (const float*)d_in[3];
    const unsigned char* mask = (const unsigned char*)d_in[4];
    const float* ln_q_g = (const float*)d_in[5];
    const float* ln_q_b = (const float*)d_in[6];
    const float* wq     = (const float*)d_in[7];
    const float* bq     = (const float*)d_in[8];
    const float* ln_k_g = (const float*)d_in[9];
    const float* ln_k_b = (const float*)d_in[10];
    const float* wk     = (const float*)d_in[11];
    const float* bk     = (const float*)d_in[12];
    const float* ln_v_g = (const float*)d_in[13];
    const float* ln_v_b = (const float*)d_in[14];
    const float* wv     = (const float*)d_in[15];
    const float* bv     = (const float*)d_in[16];
    const float* w_proj = (const float*)d_in[17];
    const float* b_proj = (const float*)d_in[18];
    const float* ln_pre_g = (const float*)d_in[19];
    const float* ln_pre_b = (const float*)d_in[20];
    const float* w_mlp1 = (const float*)d_in[21];
    const float* b_mlp1 = (const float*)d_in[22];
    const float* w_mlp2 = (const float*)d_in[23];
    const float* b_mlp2 = (const float*)d_in[24];
    const float* ln_post_g = (const float*)d_in[25];
    const float* ln_post_b = (const float*)d_in[26];
    float* out = (float*)d_out;

    char* ws = (char*)d_ws;
    size_t off = 0;
    auto alloc = [&](size_t bytes) { size_t o = off; off += (bytes + 255) & ~(size_t)255; return o; };
    float*     qx    = (float*)(ws + alloc((size_t)CN * CQ * CDIM * 4));       // 25.2 MB
    float*     skipT = (float*)(ws + alloc((size_t)CQ * CDIM * 4));            // 4.2 MB
    _Float16*  wqh   = (_Float16*)(ws + alloc((size_t)CDIM * CDIM * 2));
    _Float16*  wkh   = (_Float16*)(ws + alloc((size_t)CDIM * CDIM * 2));
    _Float16*  wvh   = (_Float16*)(ws + alloc((size_t)CDIM * CDIM * 2));
    _Float16*  wph   = (_Float16*)(ws + alloc((size_t)CDIM * CDIM * 2));
    _Float16*  w1h   = (_Float16*)(ws + alloc((size_t)CDIM * 2 * CDIM * 2));
    _Float16*  w2h   = (_Float16*)(ws + alloc((size_t)2 * CDIM * CDIM * 2));
    _Float16*  qph   = (_Float16*)(ws + alloc((size_t)CN * CQ * CDIM * 2));    // 12.6 MB
    _Float16*  kph   = (_Float16*)(ws + alloc((size_t)CN * CQ * CG * CDIM * 2)); // 100.7 MB
    _Float16*  vph   = (_Float16*)(ws + alloc((size_t)CN * CQ * CG * CDIM * 2)); // 100.7 MB
    float*     aO    = (float*)(ws + alloc((size_t)CQ * CDIM * 4));            // 4.2 MB
    float*     z     = (float*)(ws + alloc((size_t)CQ * CDIM * 4));            // 4.2 MB
    float*     h1    = (float*)(ws + alloc((size_t)CQ * 2 * CDIM * 4));        // 8.4 MB

    // 1) transpose q frames + skip to q-major
    kT_transpose<<<7 * 8 * 128, 256, 0, stream>>>(q, skip, qx, skipT);

    // 2) pre-swizzle weights to f16 B-fragment order
    kW_swizzle<<<16, 256, 0, stream>>>(wq,     wqh, 256, 256);
    kW_swizzle<<<16, 256, 0, stream>>>(wk,     wkh, 256, 256);
    kW_swizzle<<<16, 256, 0, stream>>>(wv,     wvh, 256, 256);
    kW_swizzle<<<16, 256, 0, stream>>>(w_proj, wph, 256, 256);
    kW_swizzle<<<32, 256, 0, stream>>>(w_mlp1, w1h, 256, 512);
    kW_swizzle<<<32, 256, 0, stream>>>(w_mlp2, w2h, 512, 256);

    // 3) fused LN + projection GEMMs (f16 WMMA, f32 accumulate; NT loads on
    //    the big single-use f32 streams to keep L2 free for kp/vp/weights)
    kGemm<256, true, EP_F16, true><<<(CN * CQ) / 32, 256, 0, stream>>>(
        qx, ln_q_g, ln_q_b, wqh, bq, 256, qph, nullptr, nullptr, nullptr, nullptr);
    kGemm<256, true, EP_F16, true><<<(CN * CQ * CG) / 32, 256, 0, stream>>>(
        k, ln_k_g, ln_k_b, wkh, bk, 256, kph, nullptr, nullptr, nullptr, nullptr);
    kGemm<256, true, EP_F16, true><<<(CN * CQ * CG) / 32, 256, 0, stream>>>(
        v, ln_v_g, ln_v_b, wvh, bv, 256, vph, nullptr, nullptr, nullptr, nullptr);

    // 4) attention (48 keys per (qpos, head))
    kAttn<<<(CQ * CHEADS) / 8, 256, 0, stream>>>(qph, kph, vph, mask, aO);

    // 5) out-proj + skip + LN_pre -> z
    kGemm<256, false, EP_SKIP_LN, false><<<CQ / 32, 256, 0, stream>>>(
        aO, nullptr, nullptr, wph, b_proj, 256, nullptr, skipT, ln_pre_g, ln_pre_b, z);

    // 6) MLP1 + exact GELU -> h1 (N=512 via grid.y)
    kGemm<256, false, EP_GELU_F32, false><<<dim3(CQ / 32, 2), 256, 0, stream>>>(
        z, nullptr, nullptr, w1h, b_mlp1, 512, nullptr, nullptr, nullptr, nullptr, h1);

    // 7) MLP2 + residual(z) + LN_post + transposed store -> out (256, 64, 64)
    kGemm<512, false, EP_RESID_LN_T, false><<<CQ / 32, 256, 0, stream>>>(
        h1, nullptr, nullptr, w2h, b_mlp2, 256, nullptr, z, ln_post_g, ln_post_b, out);

    (void)in_sizes; (void)n_in; (void)out_size; (void)ws_size;
}
